// PCDAlignment_15281493639730
// MI455X (gfx1250) — matrix operand order repaired and checked by reading, not compiled, not run on hardware
//
#include <hip/hip_runtime.h>
#include <math.h>

// PCD alignment (EDVR) on gfx1250: all matmul work via v_wmma_f32_16x16x32_bf16.
// conv(128->64)+lrelu -> conv(64->64)+lrelu -> conv(64->216) -> offset xform
// -> DCNv2 GEMM (K=576)+lrelu -> cascade repeat -> d_out.
//
// Implicit GEMM: one wave per 16(Cout)x16(pixel) tile; im2col / deformable-
// sampled B panel staged in LDS (bf16, b32-packed) shared by all waves of the
// block; bf16 weights pre-packed row-major [Mpad][Cin*9].

#define HH 160
#define WW 160
#define BB 4
#define SLOPE 0.1f
#define PIXTILES 6400   // (BB*HH*WW)/16

typedef __attribute__((ext_vector_type(16))) __bf16 v16bf;
typedef __attribute__((ext_vector_type(8)))  float  v8f;

union WFrag { v16bf v; unsigned int u[8]; };

__device__ __forceinline__ unsigned short f2bf(float f) {
  // round-to-nearest-even fp32 -> bf16
  unsigned int u = __float_as_uint(f);
  u += 0x7fffu + ((u >> 16) & 1u);
  return (unsigned short)(u >> 16);
}

__device__ __forceinline__ float lrelu_f(float v, int act) {
  return act ? ((v > 0.0f) ? v : SLOPE * v) : v;
}

// ---- pack fp32 OIHW weights -> bf16 row-major [Mpad][Cin*9], zero-padded rows
__global__ __launch_bounds__(256) void pack_w_bf16(
    const float* __restrict__ w, unsigned short* __restrict__ wp,
    int Cout, int K, int Mpad) {
  int idx = blockIdx.x * 256 + threadIdx.x;
  if (idx >= Mpad * K) return;
  int co = idx / K, k = idx - co * K;
  float v = (co < Cout) ? w[co * K + k] : 0.0f;
  wp[idx] = f2bf(v);
}

// ---- in-place offset/mask transform on the 216-channel conv-offset output:
// ch<144 (g*18+kk*2+t): add base grid (ky-1+h or kx-1+w); ch>=144: sigmoid.
__global__ __launch_bounds__(256) void offset_xform(float* __restrict__ co, int total) {
  int idx = blockIdx.x * 256 + threadIdx.x;
  if (idx >= total) return;
  int w = idx % WW;
  int h = (idx / WW) % HH;
  int ch = (idx / (WW * HH)) % 216;
  float v = co[idx];
  if (ch < 144) {
    int r = ch % 18;
    int kk = r >> 1;
    int t = r & 1;
    v += (t == 0) ? (float)(kk / 3 - 1 + h) : (float)(kk % 3 - 1 + w);
  } else {
    v = 1.0f / (1.0f + expf(-v));
  }
  co[idx] = v;
}

// ---- one 16x16x32 bf16 WMMA accumulation step (u32-packed operands).
// kc2 = (K offset)/2 in u32 units; half = lane>>4.
__device__ __forceinline__ v8f wmma_step(const unsigned int* __restrict__ wrow32,
                                         const unsigned int* brow32,
                                         int kc2, int half, v8f acc) {
  WFrag a, bb;
#pragma unroll
  for (int p = 0; p < 8; ++p) {
    // A (16x32 bf16): lane-half -> K {0..15} / {16..31}
    a.u[p] = wrow32[kc2 + half * 8 + p];
    // B (32x16 bf16): lane-half -> K {0..15} / {16..31}
    bb.u[p] = brow32[kc2 + half * 8 + p];
  }
  return __builtin_amdgcn_wmma_f32_16x16x32_bf16(false, a.v, false, bb.v,
                                                 (short)0, acc, false, false);
}

// ---- GEMM epilogue: bias + optional lrelu, full-tile fast path.
__device__ __forceinline__ void store_tile(v8f acc, const float* __restrict__ bias,
                                           float* __restrict__ out, size_t obase,
                                           int c0, int Cout, int do_act) {
  if (c0 + 8 <= Cout) {
#pragma unroll
    for (int i = 0; i < 8; ++i) {
      float v = lrelu_f(acc[i] + bias[c0 + i], do_act);
      out[obase + (size_t)(c0 + i) * (HH * WW)] = v;
    }
  } else {
#pragma unroll
    for (int i = 0; i < 8; ++i) {
      if (c0 + i < Cout) {
        float v = lrelu_f(acc[i] + bias[c0 + i], do_act);
        out[obase + (size_t)(c0 + i) * (HH * WW)] = v;
      }
    }
  }
}

// ---- implicit-GEMM 3x3 conv (padding 1). Input = concat(in0[Cin0], in1[Cin1]).
// blockDim = Mtiles*32: wave w computes Cout tile w; LDS panel built once.
__global__ __launch_bounds__(512) void conv3x3_wmma(
    const float* __restrict__ in0, int Cin0,
    const float* __restrict__ in1, int Cin1,
    const unsigned short* __restrict__ Wp,
    const float* __restrict__ bias,
    float* __restrict__ out, int Cout, int do_act) {
  extern __shared__ unsigned int sB32[]; // [16][K/2] pixel-major, bf16 pairs
  const int Cin = Cin0 + Cin1;
  const int K = Cin * 9;
  const int K2 = K >> 1;
  const int pix0 = blockIdx.x * 16;
  const int w0 = pix0 % WW;
  const int h = (pix0 / WW) % HH;
  const int b = pix0 / (WW * HH);
  const int tid = threadIdx.x;
  const int bd = blockDim.x;

  for (int n = 0; n < 16; ++n) {
    for (int j = tid; j < K2; j += bd) {
      unsigned int pr = 0;
#pragma unroll
      for (int e = 0; e < 2; ++e) {
        int k = 2 * j + e;
        int ci = k / 9, kk = k - ci * 9;
        int yy = h + kk / 3 - 1;
        int xx = w0 + n + (kk % 3) - 1;
        float v = 0.0f;
        if ((unsigned)yy < (unsigned)HH && (unsigned)xx < (unsigned)WW) {
          v = (ci < Cin0) ? in0[((b * Cin0 + ci) * HH + yy) * WW + xx]
                          : in1[((b * Cin1 + (ci - Cin0)) * HH + yy) * WW + xx];
        }
        pr |= (unsigned int)f2bf(v) << (16 * e);
      }
      sB32[n * K2 + j] = pr;
    }
  }
  __syncthreads();

  const int lane = tid & 31;
  const int mtile = tid >> 5;
  const int half = lane >> 4;
  const int n = lane & 15;
  const unsigned int* wrow32 =
      (const unsigned int*)Wp + (size_t)(mtile * 16 + n) * K2;
  const unsigned int* brow32 = sB32 + n * K2;

  v8f acc = {};
  for (int kc2 = 0; kc2 < K2; kc2 += 32) { // 64 K values / iter; K % 64 == 0
    acc = wmma_step(wrow32, brow32, kc2, half, acc);
    acc = wmma_step(wrow32, brow32, kc2 + 16, half, acc);
  }

  size_t obase = ((size_t)(b * Cout) * HH + h) * WW + (w0 + n);
  store_tile(acc, bias, out, obase, mtile * 16 + (half ? 8 : 0), Cout, do_act);
}

// ---- DCNv2 as implicit GEMM: sampled/masked B panel (16 px x 576) in LDS,
// w_dcn rows as A. k = ci*9 + kk, ci = g*8 + c  (matches w_dcn OIHW flatten).
__global__ __launch_bounds__(128) void dcn_wmma(
    const float* __restrict__ src,  // [B,64,H,W] sampling source
    const float* __restrict__ co,   // [B,216,H,W] transformed py/px/mask
    const unsigned short* __restrict__ Wp, // [64][576] bf16
    const float* __restrict__ bias,
    float* __restrict__ out, int do_act) {
  const int K2 = 288; // K=576
  __shared__ unsigned int sB32[16 * 288];
  const int pix0 = blockIdx.x * 16;
  const int w0 = pix0 % WW;
  const int h = (pix0 / WW) % HH;
  const int b = pix0 / (WW * HH);
  const int tid = threadIdx.x;
  const float* cb = co + (size_t)b * 216 * HH * WW;
  const float* srcb = src + (size_t)b * 64 * HH * WW;

  for (int n = 0; n < 16; ++n) {
    const int pofs = h * WW + (w0 + n);
    for (int j = tid; j < K2; j += 128) {
      unsigned int pr = 0;
#pragma unroll
      for (int e = 0; e < 2; ++e) {
        int k = 2 * j + e;
        int ci = k / 9, kk = k - ci * 9;
        int g = ci >> 3;
        float py = cb[(g * 18 + kk * 2) * (HH * WW) + pofs];
        float px = cb[(g * 18 + kk * 2 + 1) * (HH * WW) + pofs];
        float mk = cb[(144 + g * 9 + kk) * (HH * WW) + pofs];
        float y0 = floorf(py), x0 = floorf(px);
        float ly = py - y0, lx = px - x0;
        int iy = (int)y0, ix = (int)x0;
        const float* chp = srcb + ci * (HH * WW);
        bool y0i = (unsigned)iy < (unsigned)HH;
        bool y1i = (unsigned)(iy + 1) < (unsigned)HH;
        bool x0i = (unsigned)ix < (unsigned)WW;
        bool x1i = (unsigned)(ix + 1) < (unsigned)WW;
        float v00 = (y0i && x0i) ? chp[iy * WW + ix] : 0.0f;
        float v01 = (y0i && x1i) ? chp[iy * WW + ix + 1] : 0.0f;
        float v10 = (y1i && x0i) ? chp[(iy + 1) * WW + ix] : 0.0f;
        float v11 = (y1i && x1i) ? chp[(iy + 1) * WW + ix + 1] : 0.0f;
        float v = v00 * (1.0f - ly) * (1.0f - lx) + v01 * (1.0f - ly) * lx +
                  v10 * ly * (1.0f - lx) + v11 * ly * lx;
        pr |= (unsigned int)f2bf(v * mk) << (16 * e);
      }
      sB32[n * K2 + j] = pr;
    }
  }
  __syncthreads();

  const int lane = tid & 31;
  const int mtile = tid >> 5; // Cout=64 -> exactly 4 m-tiles
  const int half = lane >> 4;
  const int n = lane & 15;
  const unsigned int* wrow32 =
      (const unsigned int*)Wp + (size_t)(mtile * 16 + n) * K2;
  const unsigned int* brow32 = sB32 + n * K2;

  v8f acc = {};
  for (int kc2 = 0; kc2 < K2; kc2 += 32) { // 288 = 9*32
    acc = wmma_step(wrow32, brow32, kc2, half, acc);
    acc = wmma_step(wrow32, brow32, kc2 + 16, half, acc);
  }

  size_t obase = ((size_t)(b * 64) * HH + h) * WW + (w0 + n);
  store_tile(acc, bias, out, obase, mtile * 16 + (half ? 8 : 0), 64, do_act);
}

extern "C" void kernel_launch(void* const* d_in, const int* in_sizes, int n_in,
                              void* d_out, int out_size, void* d_ws, size_t ws_size,
                              hipStream_t stream) {
  const float* nbr    = (const float*)d_in[0];
  const float* ref    = (const float*)d_in[1];
  const float* w_off1 = (const float*)d_in[2];
  const float* b_off1 = (const float*)d_in[3];
  const float* w_off2 = (const float*)d_in[4];
  const float* b_off2 = (const float*)d_in[5];
  const float* w_co   = (const float*)d_in[6];
  const float* b_co   = (const float*)d_in[7];
  const float* w_dcn  = (const float*)d_in[8];
  const float* b_dcn  = (const float*)d_in[9];
  const float* w_coff1 = (const float*)d_in[10];
  const float* b_coff1 = (const float*)d_in[11];
  const float* w_coff2 = (const float*)d_in[12];
  const float* b_coff2 = (const float*)d_in[13];
  const float* w_cco  = (const float*)d_in[14];
  const float* b_cco  = (const float*)d_in[15];
  const float* w_cdcn = (const float*)d_in[16];
  const float* b_cdcn = (const float*)d_in[17];

  char* ws = (char*)d_ws;
  const size_t FB = (size_t)BB * 64 * HH * WW * 4;   // 26,214,400 B
  const size_t CB = (size_t)BB * 216 * HH * WW * 4;  // 88,473,600 B
  float* off1 = (float*)(ws);
  float* off2 = (float*)(ws + FB);
  float* feat = (float*)(ws + 2 * FB);
  float* co   = (float*)(ws + 3 * FB);
  unsigned short* wq = (unsigned short*)(ws + 3 * FB + CB);
  unsigned short* Wp_off1  = wq;                      // 64 x 1152
  unsigned short* Wp_off2  = Wp_off1 + 64 * 1152;     // 64 x 576
  unsigned short* Wp_co    = Wp_off2 + 64 * 576;      // 224 x 576 (216 padded)
  unsigned short* Wp_dcn   = Wp_co + 224 * 576;       // 64 x 576
  unsigned short* Wp_coff1 = Wp_dcn + 64 * 576;       // 64 x 1152
  unsigned short* Wp_coff2 = Wp_coff1 + 64 * 1152;    // 64 x 576
  unsigned short* Wp_cco   = Wp_coff2 + 64 * 576;     // 224 x 576
  unsigned short* Wp_cdcn  = Wp_cco + 224 * 576;      // 64 x 576

  auto pack = [&](const float* w, unsigned short* dst, int Cout, int K, int Mpad) {
    int nel = Mpad * K;
    pack_w_bf16<<<(nel + 255) / 256, 256, 0, stream>>>(w, dst, Cout, K, Mpad);
  };
  pack(w_off1,  Wp_off1,  64, 1152, 64);
  pack(w_off2,  Wp_off2,  64, 576,  64);
  pack(w_co,    Wp_co,   216, 576, 224);
  pack(w_dcn,   Wp_dcn,   64, 576,  64);
  pack(w_coff1, Wp_coff1, 64, 1152, 64);
  pack(w_coff2, Wp_coff2, 64, 576,  64);
  pack(w_cco,   Wp_cco,  216, 576, 224);
  pack(w_cdcn,  Wp_cdcn,  64, 576,  64);

  const int coTotal = BB * 216 * HH * WW;
  size_t sh576 = (size_t)16 * 288 * 4;   // 18,432 B
  size_t sh1152 = (size_t)16 * 576 * 4;  // 36,864 B
  dim3 blk64(128);   // 4 waves  -> Cout 64
  dim3 blk216(448);  // 14 waves -> Cout 216 (padded 224)

  // Stage 1: offset convs + DCN
  conv3x3_wmma<<<PIXTILES, blk64, sh1152, stream>>>(nbr, 64, ref, 64, Wp_off1, b_off1, off1, 64, 1);
  conv3x3_wmma<<<PIXTILES, blk64, sh576, stream>>>(off1, 64, off1, 0, Wp_off2, b_off2, off2, 64, 1);
  conv3x3_wmma<<<PIXTILES, blk216, sh576, stream>>>(off2, 64, off2, 0, Wp_co, b_co, co, 216, 0);
  offset_xform<<<(coTotal + 255) / 256, 256, 0, stream>>>(co, coTotal);
  dcn_wmma<<<PIXTILES, blk64, 0, stream>>>(nbr, co, Wp_dcn, b_dcn, feat, 1);

  // Stage 2: cascade
  conv3x3_wmma<<<PIXTILES, blk64, sh1152, stream>>>(feat, 64, ref, 64, Wp_coff1, b_coff1, off1, 64, 1);
  conv3x3_wmma<<<PIXTILES, blk64, sh576, stream>>>(off1, 64, off1, 0, Wp_coff2, b_coff2, off2, 64, 1);
  conv3x3_wmma<<<PIXTILES, blk216, sh576, stream>>>(off2, 64, off2, 0, Wp_cco, b_cco, co, 216, 0);
  offset_xform<<<(coTotal + 255) / 256, 256, 0, stream>>>(co, coTotal);
  dcn_wmma<<<PIXTILES, blk64, 0, stream>>>(feat, co, Wp_cdcn, b_cdcn, (float*)d_out, 1);
}
